// LSTM_18373870092346
// MI455X (gfx1250) — matrix operand order repaired and checked by reading, not compiled, not run on hardware
//
// MI455X (gfx1250) fused 2-layer LSTM + FC, bf16 WMMA with f32 accumulation.
//
// Design (compile-only; reasoned from MI455X specs):
//  * Batch rows are independent across the scan -> 16 WGs x 16 rows, whole
//    T=512 scan of both layers + FC fused in one persistent kernel. No global
//    sync, no giant xg scratch (only ~1.9 MB bf16 weights in d_ws).
//  * bf16 WMMA (16x16x32) = 8x FLOP/inst vs f32 WMMA (16x16x4); gates are
//    saturating so bf16 operands + f32 accumulate is the right precision knob.
//  * Weight B-fragments are re-read from the shared bf16 blob every step so
//    they stay L2-resident (1.9 MB vs 192 MB L2) and are shared by all 256
//    waves. Round-1 showed LLVM hoisting all ~120 t-invariant fragments
//    (~960 VGPRs/lane) into a scratch spill; round-2's volatile fix produced
//    SCOPE_SYS flat loads + a hard wait per load (L2 bypass!). This round
//    uses an opaque-zero (volatile s_mov into an SGPR, once per timestep) to
//    rebase the weight pointers: LICM can't hoist across iterations, but the
//    loads remain ordinary clusterable RT-scope global_load_b128 that overlap
//    with WMMAs under loadcnt.
//  * global_prefetch of the next timestep's x-tile overlaps the only
//    HBM-sourced read with the ~120 WMMAs of the current step.
//  * 16 waves/WG: wave w owns h-cols [16w,16w+16) and its i/f/g/o gate blocks;
//    c-state lives in v8f accumulator registers for the entire scan.

#include <hip/hip_runtime.h>

constexpr int T_ = 512;
constexpr int B_ = 256;
constexpr int V_ = 128;
constexpr int H_ = 256;
constexpr int G_ = 1024;   // 4*H

typedef __attribute__((ext_vector_type(16))) __bf16        v16bf;
typedef __attribute__((ext_vector_type(8)))  float         v8f;
typedef __attribute__((ext_vector_type(4)))  unsigned int  v4u;

union ABFrag { v16bf v; v4u u[2]; };

static __device__ __forceinline__ unsigned short f2bf(float x) {
  unsigned int u = __float_as_uint(x);
  u += 0x7FFFu + ((u >> 16) & 1u);            // round to nearest even
  return (unsigned short)(u >> 16);
}
static __device__ __forceinline__ float sigm(float x) {
  return 1.0f / (1.0f + __expf(-x));
}
static __device__ __forceinline__ float tanh_(float x) {
  float e = __expf(-2.0f * x);
  return (1.0f - e) / (1.0f + e);
}
static __device__ __forceinline__ v8f vzero() {
  v8f v = {0.f, 0.f, 0.f, 0.f, 0.f, 0.f, 0.f, 0.f};
  return v;
}
static __device__ __forceinline__ v8f wmma_bf16(v16bf a, v16bf b, v8f c) {
  // (neg_a, A, neg_b, B, c_mod, C, reuse_a, reuse_b)
  return __builtin_amdgcn_wmma_f32_16x16x32_bf16(false, a, false, b, (short)0, c,
                                                 false, false);
}

// Opaque zero in an SGPR: volatile asm executes once per loop iteration and
// its result is unknown to the optimizer, so pointers rebased by it are
// loop-variant (defeats LICM) while the dependent loads stay ordinary
// RT-scope global loads with normal loadcnt overlap.
static __device__ __forceinline__ int opaque_zero() {
  int v;
  asm volatile("s_mov_b32 %0, 0" : "=s"(v));
  return v;
}

// A fragment (16x32 bf16, rows striped in-lane) from row-major LDS [16][K].
// Lane L: m=L&15, half=L>>4; K = k0+16*half .. k0+16*half+15 (contiguous).
static __device__ __forceinline__ v16bf load_a(const unsigned short* lds, int K,
                                               int k0, int lane) {
  int m = lane & 15, sel = lane >> 4;
  int b0 = k0 + sel * 16;
  ABFrag f;
  f.u[0] = *(const v4u*)(lds + m * K + b0);
  f.u[1] = *(const v4u*)(lds + m * K + b0 + 8);
  return f.v;
}
// B fragment (32x16 bf16): lane n (0..15 / 16..31) holds 16 consecutive K
// values of output column n -> one contiguous 32-byte read of weight row
// (row_base+n) in the row-major bf16 weight matrix [rows][K].
static __device__ __forceinline__ v16bf load_b(const unsigned short* W,
                                               int K, int row_base, int k0, int lane) {
  int n = lane & 15, sel = lane >> 4;
  const unsigned short* p = W + (size_t)(row_base + n) * K + k0 + sel * 16;
  const v4u* q = (const v4u*)p;
  ABFrag f;
  f.u[0] = q[0];
  f.u[1] = q[1];
  return f.v;
}

// ---------------------------------------------------------------- prep kernel
__global__ void lstm_prep(const float* __restrict__ Wih0, const float* __restrict__ Whh0,
                          const float* __restrict__ Wih1, const float* __restrict__ Whh1,
                          const float* __restrict__ Wfc,
                          const float* __restrict__ bih0, const float* __restrict__ bhh0,
                          const float* __restrict__ bih1, const float* __restrict__ bhh1,
                          unsigned short* __restrict__ wsu,
                          float* __restrict__ bias0, float* __restrict__ bias1) {
  const int N0 = G_ * V_;        // 131072
  const int N1 = G_ * H_;        // 262144
  const int NF = V_ * H_;        // 32768
  const int total = N0 + 3 * N1 + NF;
  const int stride = gridDim.x * blockDim.x;
  for (int i = blockIdx.x * blockDim.x + threadIdx.x; i < total + 2 * G_; i += stride) {
    int idx = i;
    if (idx < N0) { wsu[idx] = f2bf(Wih0[idx]); continue; }
    idx -= N0;
    if (idx < N1) { wsu[N0 + idx] = f2bf(Whh0[idx]); continue; }
    idx -= N1;
    if (idx < N1) { wsu[N0 + N1 + idx] = f2bf(Wih1[idx]); continue; }
    idx -= N1;
    if (idx < N1) { wsu[N0 + 2 * N1 + idx] = f2bf(Whh1[idx]); continue; }
    idx -= N1;
    if (idx < NF) { wsu[N0 + 3 * N1 + idx] = f2bf(Wfc[idx]); continue; }
    idx -= NF;
    if (idx < G_) { bias0[idx] = bih0[idx] + bhh0[idx]; continue; }
    idx -= G_;
    bias1[idx] = bih1[idx] + bhh1[idx];
  }
}

// ---------------------------------------------------------------- main kernel
__global__ void __launch_bounds__(512, 1)
lstm_fused(const float* __restrict__ x, const float* __restrict__ h0in,
           const float* __restrict__ c0in,
           const unsigned short* __restrict__ Wih0b, const unsigned short* __restrict__ Whh0b,
           const unsigned short* __restrict__ Wih1b, const unsigned short* __restrict__ Whh1b,
           const unsigned short* __restrict__ Wfcb,
           const float* __restrict__ bias0, const float* __restrict__ bias1,
           const float* __restrict__ bfc, float* __restrict__ out) {
  __shared__ __align__(16) unsigned short xt[16 * V_];   // x_t tile, bf16
  __shared__ __align__(16) unsigned short h0l[16 * H_];  // h(layer0), bf16
  __shared__ __align__(16) unsigned short h1l[16 * H_];  // h(layer1), bf16

  const int th   = threadIdx.x;
  const int lane = th & 31;
  const int w    = th >> 5;          // wave 0..15
  const int wg   = blockIdx.x;       // batch rows [16*wg, 16*wg+16)
  const int nl   = lane & 15;
  const int sel  = lane >> 4;
  const int colb = w * 16;           // this wave's h-column block
  const int ncol = colb + nl;

  // Per-lane gate biases (column-dependent only).
  const float bi0 = bias0[ncol],       bf0 = bias0[256 + ncol];
  const float bg0 = bias0[512 + ncol], bo0 = bias0[768 + ncol];
  const float bi1 = bias1[ncol],       bf1 = bias1[256 + ncol];
  const float bg1 = bias1[512 + ncol], bo1 = bias1[768 + ncol];
  float bfcn = 0.f;
  if (w < 8) bfcn = bfc[ncol];

  // Persistent per-wave state tiles (C/D layout: lane=N, VGPR r -> M=r+8*sel).
  v8f c0a = vzero(), c1a = vzero(), h0a = vzero(), h1a = vzero();
#pragma unroll
  for (int r = 0; r < 8; ++r) {
    const int b = wg * 16 + r + 8 * sel;
    h0a[r] = h0in[(size_t)b * H_ + ncol];
    h1a[r] = h0in[(size_t)B_ * H_ + (size_t)b * H_ + ncol];
    c0a[r] = c0in[(size_t)b * H_ + ncol];
    c1a[r] = c0in[(size_t)B_ * H_ + (size_t)b * H_ + ncol];
    h0l[(r + 8 * sel) * H_ + ncol] = f2bf(h0a[r]);
    h1l[(r + 8 * sel) * H_ + ncol] = f2bf(h1a[r]);
  }
  __syncthreads();

#pragma unroll 1
  for (int t = 0; t < T_; ++t) {
    // Rebase weight pointers by an opaque zero so the ~120 fragment loads
    // below cannot be hoisted out of the scan loop (see header comment).
    const int oz = opaque_zero();
    const unsigned short* Wih0t = Wih0b + oz;
    const unsigned short* Whh0t = Whh0b + oz;
    const unsigned short* Wih1t = Wih1b + oz;
    const unsigned short* Whh1t = Whh1b + oz;
    const unsigned short* Wfct  = Wfcb + oz;

    // Stage x_t [16,128] into LDS as bf16 (coalesced; 4 elems/thread), and
    // prefetch x_{t+1} into cache so the only HBM-sourced read of the next
    // step overlaps this step's ~120 WMMAs.
#pragma unroll
    for (int i = 0; i < (16 * V_) / 512; ++i) {
      const int idx = th + i * 512;
      const int row = idx >> 7, v = idx & (V_ - 1);
      const size_t gidx = ((size_t)t * B_ + wg * 16 + row) * V_ + v;
      xt[idx] = f2bf(x[gidx]);
      if (t + 1 < T_) __builtin_prefetch(&x[gidx + (size_t)B_ * V_], 0, 1);
    }
    __syncthreads();

    // -------- layer 0: g = x_t @ Wih0^T + h0 @ Whh0^T + bias0
    v8f ai = vzero(), af = vzero(), ag = vzero(), ao = vzero();
#pragma unroll
    for (int k0 = 0; k0 < V_; k0 += 32) {
      const v16bf a = load_a(xt, V_, k0, lane);
      ai = wmma_bf16(a, load_b(Wih0t, V_, 0 + colb, k0, lane), ai);
      af = wmma_bf16(a, load_b(Wih0t, V_, 256 + colb, k0, lane), af);
      ag = wmma_bf16(a, load_b(Wih0t, V_, 512 + colb, k0, lane), ag);
      ao = wmma_bf16(a, load_b(Wih0t, V_, 768 + colb, k0, lane), ao);
    }
#pragma unroll
    for (int k0 = 0; k0 < H_; k0 += 32) {
      const v16bf a = load_a(h0l, H_, k0, lane);
      ai = wmma_bf16(a, load_b(Whh0t, H_, 0 + colb, k0, lane), ai);
      af = wmma_bf16(a, load_b(Whh0t, H_, 256 + colb, k0, lane), af);
      ag = wmma_bf16(a, load_b(Whh0t, H_, 512 + colb, k0, lane), ag);
      ao = wmma_bf16(a, load_b(Whh0t, H_, 768 + colb, k0, lane), ao);
    }
#pragma unroll
    for (int r = 0; r < 8; ++r) {
      const float iv = sigm(ai[r] + bi0), fv = sigm(af[r] + bf0);
      const float gv = tanh_(ag[r] + bg0), ov = sigm(ao[r] + bo0);
      const float c = fv * c0a[r] + iv * gv;
      c0a[r] = c;
      h0a[r] = ov * tanh_(c);
    }
    __syncthreads();   // all waves done reading h0l(t-1)
#pragma unroll
    for (int r = 0; r < 8; ++r) h0l[(r + 8 * sel) * H_ + ncol] = f2bf(h0a[r]);
    __syncthreads();

    // -------- layer 1: g = h0_t @ Wih1^T + h1 @ Whh1^T + bias1
    ai = vzero(); af = vzero(); ag = vzero(); ao = vzero();
#pragma unroll
    for (int k0 = 0; k0 < H_; k0 += 32) {
      const v16bf a = load_a(h0l, H_, k0, lane);
      ai = wmma_bf16(a, load_b(Wih1t, H_, 0 + colb, k0, lane), ai);
      af = wmma_bf16(a, load_b(Wih1t, H_, 256 + colb, k0, lane), af);
      ag = wmma_bf16(a, load_b(Wih1t, H_, 512 + colb, k0, lane), ag);
      ao = wmma_bf16(a, load_b(Wih1t, H_, 768 + colb, k0, lane), ao);
    }
#pragma unroll
    for (int k0 = 0; k0 < H_; k0 += 32) {
      const v16bf a = load_a(h1l, H_, k0, lane);
      ai = wmma_bf16(a, load_b(Whh1t, H_, 0 + colb, k0, lane), ai);
      af = wmma_bf16(a, load_b(Whh1t, H_, 256 + colb, k0, lane), af);
      ag = wmma_bf16(a, load_b(Whh1t, H_, 512 + colb, k0, lane), ag);
      ao = wmma_bf16(a, load_b(Whh1t, H_, 768 + colb, k0, lane), ao);
    }
#pragma unroll
    for (int r = 0; r < 8; ++r) {
      const float iv = sigm(ai[r] + bi1), fv = sigm(af[r] + bf1);
      const float gv = tanh_(ag[r] + bg1), ov = sigm(ao[r] + bo1);
      const float c = fv * c1a[r] + iv * gv;
      c1a[r] = c;
      h1a[r] = ov * tanh_(c);
    }
    __syncthreads();
#pragma unroll
    for (int r = 0; r < 8; ++r) h1l[(r + 8 * sel) * H_ + ncol] = f2bf(h1a[r]);
    __syncthreads();

    // -------- FC: out_t = h1_t @ Wfc^T + bfc   (waves 0..7, wave-uniform)
    if (w < 8) {
      v8f acc = vzero();
#pragma unroll
      for (int k0 = 0; k0 < H_; k0 += 32)
        acc = wmma_bf16(load_a(h1l, H_, k0, lane),
                        load_b(Wfct, H_, colb, k0, lane), acc);
#pragma unroll
      for (int r = 0; r < 8; ++r) {
        const int b = wg * 16 + r + 8 * sel;
        out[((size_t)t * B_ + b) * V_ + ncol] = acc[r] + bfcn;
      }
    }
    __syncthreads();   // protect h1l / xt for next iteration
  }

  // Final hidden/cell state: [2,B,H] each, appended after the [T*B,V] output.
  float* hid = out + (size_t)T_ * B_ * V_;
  float* cel = hid + 2 * (size_t)B_ * H_;
#pragma unroll
  for (int r = 0; r < 8; ++r) {
    const int b = wg * 16 + r + 8 * sel;
    hid[(size_t)b * H_ + ncol] = h0a[r];
    hid[(size_t)B_ * H_ + (size_t)b * H_ + ncol] = h1a[r];
    cel[(size_t)b * H_ + ncol] = c0a[r];
    cel[(size_t)B_ * H_ + (size_t)b * H_ + ncol] = c1a[r];
  }
}

// ---------------------------------------------------------------- launcher
extern "C" void kernel_launch(void* const* d_in, const int* in_sizes, int n_in,
                              void* d_out, int out_size, void* d_ws, size_t ws_size,
                              hipStream_t stream) {
  (void)in_sizes; (void)n_in; (void)out_size; (void)ws_size;
  const float* x    = (const float*)d_in[0];
  const float* h0   = (const float*)d_in[1];
  const float* c0   = (const float*)d_in[2];
  const float* Wih0 = (const float*)d_in[3];
  const float* Whh0 = (const float*)d_in[4];
  const float* bih0 = (const float*)d_in[5];
  const float* bhh0 = (const float*)d_in[6];
  const float* Wih1 = (const float*)d_in[7];
  const float* Whh1 = (const float*)d_in[8];
  const float* bih1 = (const float*)d_in[9];
  const float* bhh1 = (const float*)d_in[10];
  const float* Wfc  = (const float*)d_in[11];
  const float* bfc  = (const float*)d_in[12];
  float* out = (float*)d_out;

  // ws layout (bf16 weights, then f32 fused biases): ~1.91 MB total.
  unsigned short* wsu = (unsigned short*)d_ws;
  const size_t NW = (size_t)G_ * V_ + 3 * (size_t)G_ * H_ + (size_t)V_ * H_;  // 950272
  float* bias0 = (float*)((char*)d_ws + NW * 2);
  float* bias1 = bias0 + G_;

  lstm_prep<<<1024, 256, 0, stream>>>(Wih0, Whh0, Wih1, Whh1, Wfc,
                                      bih0, bhh0, bih1, bhh1, wsu, bias0, bias1);

  const unsigned short* Wih0b = wsu;
  const unsigned short* Whh0b = wsu + (size_t)G_ * V_;
  const unsigned short* Wih1b = Whh0b + (size_t)G_ * H_;
  const unsigned short* Whh1b = Wih1b + (size_t)G_ * H_;
  const unsigned short* Wfcb  = Whh1b + (size_t)G_ * H_;

  lstm_fused<<<B_ / 16, 512, 0, stream>>>(x, h0, c0, Wih0b, Whh0b, Wih1b, Whh1b,
                                          Wfcb, bias0, bias1, bfc, out);
}
